// Attention_9070970929715
// MI455X (gfx1250) — matrix operand order repaired and checked by reading, not compiled, not run on hardware
//
#include <hip/hip_runtime.h>

typedef __attribute__((ext_vector_type(16))) _Float16 v16h;
typedef __attribute__((ext_vector_type(8)))  _Float16 v8h;
typedef __attribute__((ext_vector_type(8)))  float    v8f;
typedef int v4i __attribute__((vector_size(16)));

#define HAS_ASYNC_LDS __has_builtin(__builtin_amdgcn_global_load_async_to_lds_b128)

// 16-byte global -> LDS copy. Prefers CDNA5 async-to-LDS (ASYNCcnt, no VGPR
// round trip); falls back to load+store if the builtin is unavailable.
__device__ __forceinline__ void cp16B(const _Float16* g, _Float16* l) {
#if HAS_ASYNC_LDS
  __builtin_amdgcn_global_load_async_to_lds_b128(
      (__attribute__((address_space(1))) v4i*)g,
      (__attribute__((address_space(3))) v4i*)l, 0, 0);
#else
  *(v8h*)l = *(const v8h*)g;
#endif
}

__device__ __forceinline__ void async_wait() {
#if HAS_ASYNC_LDS
#if __has_builtin(__builtin_amdgcn_s_wait_asynccnt)
  __builtin_amdgcn_s_wait_asynccnt(0);
#else
  asm volatile("s_wait_asynccnt 0x0" ::: "memory");
#endif
#endif
}

// ---------------------------------------------------------------------------
// WMMA helpers (gfx1250, wave32). D = A(16x32 f16) x B(32x16 f16) + C(16x16 f32)
// ---------------------------------------------------------------------------
__device__ __forceinline__ v8f wmma_f(v16h a, v16h b, v8f c) {
  return __builtin_amdgcn_wmma_f32_16x16x32_f16(
      /*neg_a=*/false, a, /*neg_b=*/false, b,
      /*c_mod=*/(short)0, c, /*reuse_a=*/false, /*reuse_b=*/false);
}

__device__ __forceinline__ v16h frag_ld(const _Float16* p0, const _Float16* p1) {
  union { v16h v; v8h h[2]; } u;
  u.h[0] = *(const v8h*)p0;
  u.h[1] = *(const v8h*)p1;
  return u.v;
}

// A-matrix 16x32 fragment from a row-major [16][stride] f16 tile.
// Lane L holds row M=L&15; halves 0..7 = K=koff..koff+7,
// halves 8..15 = K=16+koff..16+koff+7, koff = (L>=16)?8:0.
__device__ __forceinline__ v16h load_a(const _Float16* base, int stride) {
  const int lane = threadIdx.x & 31;
  const int row  = lane & 15;
  const int k8   = (lane >> 4) << 3;
  const _Float16* p = base + row * stride + k8;
  return frag_ld(p, p + 16);
}

// B-matrix 32x16 fragment supplied as B^T rows: row-major [16][stride] tile
// where row = N (column of B), cols = K. Lane L holds N=L&15,
// halves 0..7 = K=koff..koff+7, halves 8..15 = K=16+koff..16+koff+7, koff = (L>=16)?8:0.
__device__ __forceinline__ v16h load_b(const _Float16* base, int stride) {
  const int lane = threadIdx.x & 31;
  const int col  = lane & 15;
  const int k16  = (lane >> 4) << 3;
  const _Float16* p = base + col * stride + k16;
  return frag_ld(p, p + 16);
}

// ---------------------------------------------------------------------------
// f32 -> f16 convert
// ---------------------------------------------------------------------------
__global__ void cvt_f32_f16(const float* __restrict__ src,
                            _Float16* __restrict__ dst, int n) {
  int i = blockIdx.x * blockDim.x + threadIdx.x;
  const int stride = gridDim.x * blockDim.x;
  for (; i < n; i += stride) dst[i] = (_Float16)src[i];
}

// ---------------------------------------------------------------------------
// Shared double-buffered GEMM mainloop: acc += A[128 rows]@W^T[128 rows],
// K=1024 in 16 steps of 64. Block 256 threads, wave (w&3, w>>2) owns 32x64.
// ---------------------------------------------------------------------------
#define GSTR 72  // 64 + 8 halves pad (144B, multiple of 16B)
#define GBUF (128 * GSTR)

__device__ __forceinline__ void gemm_tiles_load(
    const _Float16* __restrict__ A, const _Float16* __restrict__ B,
    _Float16* sA, _Float16* sB, int m0, int n0, int k0) {
  const int tid = threadIdx.x;
#pragma unroll
  for (int i = 0; i < 4; ++i) {
    const int c = tid + i * 256;          // 1024 chunks of 8 halves per tile
    const int row = c >> 3, cc = (c & 7) * 8;
    cp16B(&A[(m0 + row) * 1024 + k0 + cc], &sA[row * GSTR + cc]);
    cp16B(&B[(n0 + row) * 1024 + k0 + cc], &sB[row * GSTR + cc]);
  }
}

__device__ __forceinline__ void gemm_mainloop(
    const _Float16* __restrict__ A, const _Float16* __restrict__ B,
    _Float16* sA, _Float16* sB,       // each [2][GBUF]
    int m0, int n0, v8f acc[2][4]) {
  const int w = threadIdx.x >> 5;
  const int wm = w & 3, wn = w >> 2;

  gemm_tiles_load(A, B, sA, sB, m0, n0, 0);
  async_wait();
  __syncthreads();

  for (int kt = 0; kt < 16; ++kt) {
    const int cur = kt & 1;
    if (kt + 1 < 16) {                   // prefetch next K-tile (overlaps WMMA)
      const int nxt = 1 - cur;
      gemm_tiles_load(A, B, sA + nxt * GBUF, sB + nxt * GBUF,
                      m0, n0, (kt + 1) * 64);
    }
    const _Float16* cA = sA + cur * GBUF;
    const _Float16* cB = sB + cur * GBUF;
#pragma unroll
    for (int ks = 0; ks < 64; ks += 32) {
      const v16h a0 = load_a(&cA[(wm * 32 +  0) * GSTR + ks], GSTR);
      const v16h a1 = load_a(&cA[(wm * 32 + 16) * GSTR + ks], GSTR);
#pragma unroll
      for (int nt = 0; nt < 4; ++nt) {
        const v16h b = load_b(&cB[(wn * 64 + nt * 16) * GSTR + ks], GSTR);
        acc[0][nt] = wmma_f(a0, b, acc[0][nt]);
        acc[1][nt] = wmma_f(a1, b, acc[1][nt]);
      }
    }
    async_wait();                        // prefetch landed
    __syncthreads();                     // all waves done reading `cur`
  }
}

// ---------------------------------------------------------------------------
// QKV GEMM: y[4096][3072] = x16 @ wq16^T; scatter to q*scale / k / vT (f16).
// ---------------------------------------------------------------------------
__global__ __launch_bounds__(256) void qkv_gemm(
    const _Float16* __restrict__ x16, const _Float16* __restrict__ w16,
    _Float16* __restrict__ q16, _Float16* __restrict__ k16,
    _Float16* __restrict__ vT16) {
  __shared__ _Float16 sA[2 * GBUF];
  __shared__ _Float16 sB[2 * GBUF];
  const int lane = threadIdx.x & 31, w = threadIdx.x >> 5;
  const int wm = w & 3, wn = w >> 2;
  const int m0 = blockIdx.y * 128, n0 = blockIdx.x * 128;

  v8f acc[2][4] = {};
  gemm_mainloop(x16, w16, sA, sB, m0, n0, acc);

  const int col = lane & 15, hi8 = (lane >> 4) << 3;
#pragma unroll
  for (int mt = 0; mt < 2; ++mt)
#pragma unroll
    for (int nt = 0; nt < 4; ++nt)
#pragma unroll
      for (int r = 0; r < 8; ++r) {
        const float v = acc[mt][nt][r];
        const int m = m0 + wm * 32 + mt * 16 + hi8 + r;
        const int o = n0 + wn * 64 + nt * 16 + col;
        if (o < 1024) {
          const int h = o >> 6, d = o & 63;
          q16[(h * 4096 + m) * 64 + d] = (_Float16)(v * 0.125f);  // D^-0.5
        } else if (o < 2048) {
          const int oo = o - 1024, h = oo >> 6, d = oo & 63;
          k16[(h * 4096 + m) * 64 + d] = (_Float16)v;
        } else {
          const int oo = o - 2048, h = oo >> 6, d = oo & 63;
          vT16[(h * 64 + d) * 4096 + m] = (_Float16)v;
        }
      }
}

// ---------------------------------------------------------------------------
// Causal flash attention. Grid: (m-block 0..31, head 0..15). 8 waves/block,
// each wave owns 16 Q rows; 32-key blocks, double-buffered K/V^T in LDS.
// ---------------------------------------------------------------------------
#define KSTR 72
#define VSTR 40
#define PSTR 40
#define KBUF (32 * KSTR)
#define VBUF (64 * VSTR)

__device__ __forceinline__ void attn_tiles_load(
    const _Float16* __restrict__ k16, const _Float16* __restrict__ vT16,
    _Float16* sK, _Float16* sV, int h, int n0) {
  const int tid = threadIdx.x;
  const int kr = tid >> 3, kc = (tid & 7) * 8;     // 32x64 = 256 chunks
  cp16B(&k16[(h * 4096 + n0 + kr) * 64 + kc], &sK[kr * KSTR + kc]);
  const int vr = tid >> 2, vc = (tid & 3) * 8;     // 64x32 = 256 chunks
  cp16B(&vT16[(h * 64 + vr) * 4096 + n0 + vc], &sV[vr * VSTR + vc]);
}

__global__ __launch_bounds__(256) void flash_attn(
    const _Float16* __restrict__ q16, const _Float16* __restrict__ k16,
    const _Float16* __restrict__ vT16, _Float16* __restrict__ attn16) {
  __shared__ _Float16 sK[2 * KBUF];        // K block [32 keys][64 d]
  __shared__ _Float16 sV[2 * VBUF];        // V^T block [64 d][32 keys]
  __shared__ _Float16 sP[8 * 16 * PSTR];   // per-wave P [16 m][32 keys]
  const int lane = threadIdx.x & 31, w = threadIdx.x >> 5;
  const int h = blockIdx.y;
  const int m_base = blockIdx.x * 128;
  const int col = lane & 15, hi8 = (lane >> 4) << 3;
  const int mrow0 = m_base + w * 16;
  const int mrow  = mrow0 + hi8;           // row for VGPR r is mrow + r

  // Q tile as two A-fragments (d 0..31, d 32..63), kept in registers.
  const _Float16* qbase = q16 + (h * 4096 + mrow0) * 64;
  const v16h aQ0 = load_a(qbase, 64);
  const v16h aQ1 = load_a(qbase + 32, 64);

  v8f o0 = {}, o1 = {}, o2 = {}, o3 = {};
  float mstate[8], lstate[8];
#pragma unroll
  for (int r = 0; r < 8; ++r) { mstate[r] = -3.0e38f; lstate[r] = 0.0f; }

  const int nblocks = m_base / 32 + 4;     // keys up to m_base+127

  attn_tiles_load(k16, vT16, sK, sV, h, 0);
  async_wait();
  __syncthreads();

  for (int j = 0; j < nblocks; ++j) {
    const int n0 = j * 32;
    const int cur = j & 1;
    if (j + 1 < nblocks) {                 // prefetch next key block
      const int nxt = 1 - cur;
      attn_tiles_load(k16, vT16, sK + nxt * KBUF, sV + nxt * VBUF,
                      h, n0 + 32);
    }
    const _Float16* cK = sK + cur * KBUF;
    const _Float16* cV = sV + cur * VBUF;

    // S = Q K^T for two 16-key tiles (f32 accum; q pre-scaled)
    v8f s0 = {}, s1 = {};
    s0 = wmma_f(aQ0, load_b(&cK[ 0 * KSTR +  0], KSTR), s0);
    s1 = wmma_f(aQ0, load_b(&cK[16 * KSTR +  0], KSTR), s1);
    s0 = wmma_f(aQ1, load_b(&cK[ 0 * KSTR + 32], KSTR), s0);
    s1 = wmma_f(aQ1, load_b(&cK[16 * KSTR + 32], KSTR), s1);

    // causal mask + online softmax (rows live on 16-lane halves)
    _Float16* pw = &sP[w * 16 * PSTR];
#pragma unroll
    for (int r = 0; r < 8; ++r) {
      const int m = mrow + r;
      float v0 = (n0 + col      <= m) ? s0[r] : -3.0e38f;
      float v1 = (n0 + 16 + col <= m) ? s1[r] : -3.0e38f;
      float t = fmaxf(v0, v1);
      t = fmaxf(t, __shfl_xor(t, 1, 32));
      t = fmaxf(t, __shfl_xor(t, 2, 32));
      t = fmaxf(t, __shfl_xor(t, 4, 32));
      t = fmaxf(t, __shfl_xor(t, 8, 32));
      const float mnew  = fmaxf(mstate[r], t);
      const float scale = __expf(mstate[r] - mnew);
      const float p0 = __expf(v0 - mnew);
      const float p1 = __expf(v1 - mnew);
      float rs = p0 + p1;
      rs += __shfl_xor(rs, 1, 32);
      rs += __shfl_xor(rs, 2, 32);
      rs += __shfl_xor(rs, 4, 32);
      rs += __shfl_xor(rs, 8, 32);
      lstate[r] = lstate[r] * scale + rs;
      mstate[r] = mnew;
      o0[r] *= scale; o1[r] *= scale; o2[r] *= scale; o3[r] *= scale;
      pw[(hi8 + r) * PSTR +      col] = (_Float16)p0;   // C-layout -> LDS
      pw[(hi8 + r) * PSTR + 16 + col] = (_Float16)p1;
    }
    __syncthreads();                       // P visible

    // O += P(16x32) x V(32x16 per d-tile)
    const v16h aP = load_a(pw, PSTR);
    o0 = wmma_f(aP, load_b(&cV[ 0 * VSTR], VSTR), o0);
    o1 = wmma_f(aP, load_b(&cV[16 * VSTR], VSTR), o1);
    o2 = wmma_f(aP, load_b(&cV[32 * VSTR], VSTR), o2);
    o3 = wmma_f(aP, load_b(&cV[48 * VSTR], VSTR), o3);

    async_wait();                          // prefetch landed
    __syncthreads();                       // all waves done with `cur`
  }

  // normalize and emit attn output row-major [n][h*64+d] as f16
#pragma unroll
  for (int r = 0; r < 8; ++r) {
    const float rl = 1.0f / lstate[r];
    _Float16* dst = attn16 + (mrow + r) * 1024 + h * 64;
    dst[ 0 + col] = (_Float16)(o0[r] * rl);
    dst[16 + col] = (_Float16)(o1[r] * rl);
    dst[32 + col] = (_Float16)(o2[r] * rl);
    dst[48 + col] = (_Float16)(o3[r] * rl);
  }
}

// ---------------------------------------------------------------------------
// Output projection: out[4096][1024] = attn16 @ wp16^T + bias (f32 out)
// ---------------------------------------------------------------------------
__global__ __launch_bounds__(256) void proj_gemm(
    const _Float16* __restrict__ a16, const _Float16* __restrict__ w16,
    const float* __restrict__ bias, float* __restrict__ out) {
  __shared__ _Float16 sA[2 * GBUF];
  __shared__ _Float16 sB[2 * GBUF];
  const int lane = threadIdx.x & 31, w = threadIdx.x >> 5;
  const int wm = w & 3, wn = w >> 2;
  const int m0 = blockIdx.y * 128, n0 = blockIdx.x * 128;

  v8f acc[2][4] = {};
  gemm_mainloop(a16, w16, sA, sB, m0, n0, acc);

  const int col = lane & 15, hi8 = (lane >> 4) << 3;
#pragma unroll
  for (int mt = 0; mt < 2; ++mt)
#pragma unroll
    for (int nt = 0; nt < 4; ++nt)
#pragma unroll
      for (int r = 0; r < 8; ++r) {
        const int m = m0 + wm * 32 + mt * 16 + hi8 + r;
        const int o = n0 + wn * 64 + nt * 16 + col;
        out[m * 1024 + o] = acc[mt][nt][r] + bias[o];
      }
}

// ---------------------------------------------------------------------------
// Launcher. Workspace layout (bytes):
//   x16    @  0MB (8MB)   wq16 @  8MB (6MB)   wp16 @ 14MB (2MB)
//   q16    @ 16MB (8MB)   k16  @ 24MB (8MB)   vT16 @ 32MB (8MB)
//   attn16 @ 40MB (8MB)   total 48MB
// ---------------------------------------------------------------------------
extern "C" void kernel_launch(void* const* d_in, const int* in_sizes, int n_in,
                              void* d_out, int out_size, void* d_ws,
                              size_t ws_size, hipStream_t stream) {
  const float* x      = (const float*)d_in[0];
  const float* w_qkv  = (const float*)d_in[1];
  const float* w_proj = (const float*)d_in[2];
  const float* b_proj = (const float*)d_in[3];
  float* out = (float*)d_out;

  char* ws = (char*)d_ws;
  _Float16* x16    = (_Float16*)(ws + (size_t) 0 * (1u << 20));
  _Float16* wq16   = (_Float16*)(ws + (size_t) 8 * (1u << 20));
  _Float16* wp16   = (_Float16*)(ws + (size_t)14 * (1u << 20));
  _Float16* q16    = (_Float16*)(ws + (size_t)16 * (1u << 20));
  _Float16* k16    = (_Float16*)(ws + (size_t)24 * (1u << 20));
  _Float16* vT16   = (_Float16*)(ws + (size_t)32 * (1u << 20));
  _Float16* attn16 = (_Float16*)(ws + (size_t)40 * (1u << 20));

  cvt_f32_f16<<<2048, 256, 0, stream>>>(x,      x16,  4096 * 1024);
  cvt_f32_f16<<<2048, 256, 0, stream>>>(w_qkv,  wq16, 3072 * 1024);
  cvt_f32_f16<<<1024, 256, 0, stream>>>(w_proj, wp16, 1024 * 1024);

  qkv_gemm<<<dim3(24, 32), 256, 0, stream>>>(x16, wq16, q16, k16, vT16);
  flash_attn<<<dim3(32, 16), 256, 0, stream>>>(q16, k16, vT16, attn16);
  proj_gemm<<<dim3(8, 32), 256, 0, stream>>>(attn16, wp16, b_proj, out);
}
